// LowRankSparseAttention_24867860644391
// MI455X (gfx1250) — compile-verified, matrix operand order checked by reading, not run on hardware
//
#include <hip/hip_runtime.h>

// Problem constants (from reference)
#define BB   2
#define SS   2048
#define DM   768
#define NQK  16
#define VKV  16
#define SV   (SS + VKV)   // 2064 keys incl. virtual

typedef __bf16 bf16_t;
typedef __attribute__((ext_vector_type(16))) __bf16 v16bf;
typedef __attribute__((ext_vector_type(8)))  __bf16 v8bf;
typedef __attribute__((ext_vector_type(8)))  float  v8f;

static __device__ __forceinline__ v8bf ldv8(const bf16_t* p) {
  return *reinterpret_cast<const v8bf*>(p);
}
static __device__ __forceinline__ v16bf catv(v8bf lo, v8bf hi) {
  v16bf r;
#pragma unroll
  for (int i = 0; i < 8; ++i) { r[i] = lo[i]; r[i + 8] = hi[i]; }
  return r;
}
static __device__ __forceinline__ v16bf cat0(v8bf lo) {
  v16bf r;
#pragma unroll
  for (int i = 0; i < 8; ++i) { r[i] = lo[i]; r[i + 8] = (__bf16)0.0f; }
  return r;
}
static __device__ __forceinline__ v8f vzero8() {
  v8f r;
#pragma unroll
  for (int i = 0; i < 8; ++i) r[i] = 0.0f;
  return r;
}
static __device__ __forceinline__ v8f wmma_bf16(v16bf a, v16bf b, v8f c) {
  // D = A(16x32 bf16) * B(32x16 bf16) + C(16x16 f32)
  return __builtin_amdgcn_wmma_f32_16x16x32_bf16(false, a, false, b, (short)0, c,
                                                 false, false);
}

// ---------------- prep kernels: f32 -> bf16 packing / transposes ----------------

__global__ void k_cvt(const float* __restrict__ x, bf16_t* __restrict__ y, int n) {
  int i = blockIdx.x * blockDim.x + threadIdx.x;
  if (i < n) y[i] = (bf16_t)x[i];
}

// W_Q/W_K (h, d, e) -> wqk[c*768 + d], c = (is_k?256:0) + h*16 + e  (bf16, row-major c, contiguous d)
__global__ void k_pack_wqk(const float* __restrict__ wq, const float* __restrict__ wk,
                           bf16_t* __restrict__ out, int n) {
  int i = blockIdx.x * blockDim.x + threadIdx.x;
  if (i >= n) return;
  int c = i / DM, d = i % DM;
  float v;
  if (c < 256) v = wq[(size_t)(c >> 4) * (DM * 16) + (size_t)d * 16 + (c & 15)];
  else { int cc = c - 256; v = wk[(size_t)(cc >> 4) * (DM * 16) + (size_t)d * 16 + (cc & 15)]; }
  out[i] = (bf16_t)v;
}

// W_O (ch, 1, m) -> wot[m*256 + ch] bf16 (contiguous ch per m)
__global__ void k_pack_wo(const float* __restrict__ wo, bf16_t* __restrict__ out, int n) {
  int i = blockIdx.x * blockDim.x + threadIdx.x;
  if (i >= n) return;
  int m = i >> 8, ch = i & 255;
  out[i] = (bf16_t)wo[(size_t)ch * DM + m];
}

// virtual_k (j,h,e) -> kbuf rows S..S+15 (position-major); virtual_v (j,n) -> vt (channel-major)
__global__ void k_virtual(const float* __restrict__ vk, const float* __restrict__ vv,
                          bf16_t* __restrict__ kbuf, bf16_t* __restrict__ vt, int n) {
  int i = blockIdx.x * blockDim.x + threadIdx.x;
  if (i >= n) return;
  int b = i / (VKV * 256);
  int rem = i % (VKV * 256);
  int j = rem >> 8, c = rem & 255;
  kbuf[((size_t)(b * SV + SS + j)) * 256 + c] = (bf16_t)vk[j * 256 + c];
  vt[((size_t)(b * 256 + c)) * SV + SS + j]  = (bf16_t)vv[j * 256 + c];
}

// ---------------- QK projection + rotary: C^T = Wqk^T(A) x resid^T(B) ----------------
// one wave per (gtile, ctile): 256 gtiles x 32 ctiles = 8192 waves

__global__ void __launch_bounds__(256) k_proj_qk(const bf16_t* __restrict__ xr,
                                                 const bf16_t* __restrict__ wqk,
                                                 const float* __restrict__ bq,
                                                 const float* __restrict__ bk,
                                                 bf16_t* __restrict__ qbuf,
                                                 bf16_t* __restrict__ kbuf) {
  int wid  = (blockIdx.x * blockDim.x + threadIdx.x) >> 5;
  int lane = threadIdx.x & 31;
  int ct = wid & 31;          // column tile (0..15 -> Q, 16..31 -> K)
  int gt = wid >> 5;          // row tile over B*S
  int hl = lane >> 4, ln = lane & 15, kb = hl * 8;
  int g    = gt * 16 + ln;        // B-lane position
  int colA = ct * 16 + ln;        // A-lane weight column
  const bf16_t* wrow = wqk + (size_t)colA * DM;
  const bf16_t* xrow = xr + (size_t)g * DM;

  v8f acc = vzero8();
#pragma unroll 4
  for (int d0 = 0; d0 < DM; d0 += 32) {
    v16bf a = catv(ldv8(wrow + d0 + kb), ldv8(wrow + d0 + 16 + kb));
    v16bf b = catv(ldv8(xrow + d0 + kb), ldv8(xrow + d0 + 16 + kb));
    acc = wmma_bf16(a, b, acc);
  }

  // C layout: lane -> position g (N), vgpr r -> column ct*16 + 8*hl + r (M)
  int s = g % SS;
  int bidx = g / SS;
  float val[8];
#pragma unroll
  for (int r = 0; r < 8; ++r) {
    int c = ct * 16 + 8 * hl + r;
    float bias = (c < 256) ? bq[c] : bk[c - 256];
    val[r] = acc[r] + bias;
  }
  // rotary: e = 8*hl + r (ROT_DIM==16==D_QK); partner x[e^8] lives in lane^16, same r
#pragma unroll
  for (int r = 0; r < 8; ++r) {
    float other = __shfl_xor(val[r], 16, 32);
    int e = 8 * hl + r;
    float sgn = (e < 8) ? -1.0f : 1.0f;            // rot = [-x[8:], x[:8]]
    float inv_freq = __powf(10000.0f, -(float)r * 0.125f);  // e%8 == r
    float ang = (float)s * inv_freq;
    float sn, cs;
    __sincosf(ang, &sn, &cs);
    val[r] = val[r] * cs + sgn * other * sn;
  }
  v8bf o;
#pragma unroll
  for (int r = 0; r < 8; ++r) o[r] = (bf16_t)val[r];
  int cbase = ct * 16 + 8 * hl;
  if (ct < 16) {
    *reinterpret_cast<v8bf*>(qbuf + (size_t)g * 256 + cbase) = o;
  } else {
    *reinterpret_cast<v8bf*>(kbuf + ((size_t)bidx * SV + s) * 256 + (cbase - 256)) = o;
  }
}

// ---------------- V projection: C = resid(A) x Wv(B), stored channel-major ----------------
// 256 gtiles x 16 ctiles = 4096 waves

__global__ void __launch_bounds__(256) k_proj_v(const bf16_t* __restrict__ xr,
                                                const bf16_t* __restrict__ wv,
                                                const float* __restrict__ bv,
                                                bf16_t* __restrict__ vt) {
  int wid  = (blockIdx.x * blockDim.x + threadIdx.x) >> 5;
  int lane = threadIdx.x & 31;
  int ct = wid & 15;
  int gt = wid >> 4;
  int hl = lane >> 4, ln = lane & 15, kb = hl * 8;
  int gA = gt * 16 + ln;          // A-lane position
  int ch = ct * 16 + ln;          // B-lane channel
  const bf16_t* xrow = xr + (size_t)gA * DM;
  const bf16_t* wrow = wv + (size_t)ch * DM;

  v8f acc = vzero8();
#pragma unroll 4
  for (int d0 = 0; d0 < DM; d0 += 32) {
    v16bf a = catv(ldv8(xrow + d0 + kb), ldv8(xrow + d0 + 16 + kb));
    v16bf b = catv(ldv8(wrow + d0 + kb), ldv8(wrow + d0 + 16 + kb));
    acc = wmma_bf16(a, b, acc);
  }
  // C layout: lane -> channel ch (N), vgpr r -> position gt*16 + 8*hl + r (M)
  float bias = bv[ch];
  v8bf o;
#pragma unroll
  for (int r = 0; r < 8; ++r) o[r] = (bf16_t)(acc[r] + bias);
  int p0 = gt * 16 + 8 * hl;      // 8 consecutive positions (tiles never cross batch)
  int b_ = p0 / SS, s0 = p0 % SS;
  *reinterpret_cast<v8bf*>(vt + ((size_t)(b_ * 256 + ch)) * SV + s0) = o;
}

// ---------------- fused flash attention: S^T = K x Q^T ; O^T = V^T x P^T ----------------
// one wave per (b, h, qtile): 2*16*128 = 4096 waves

__global__ void __launch_bounds__(256) k_attn(const bf16_t* __restrict__ qbuf,
                                              const bf16_t* __restrict__ kbuf,
                                              const bf16_t* __restrict__ vt,
                                              bf16_t* __restrict__ zbuf) {
  int wid  = (blockIdx.x * blockDim.x + threadIdx.x) >> 5;
  int lane = threadIdx.x & 31;
  int b  = wid & 1;
  int h  = (wid >> 1) & 15;
  int qt = wid >> 5;                       // 0..127
  int hl = lane >> 4, ln = lane & 15, kb = hl * 8;
  int q0 = qt * 16;
  int qi = q0 + ln;                        // this lane's query index
  int g  = b * SS + qi;

  // Q^T as B fragment: lane=query, packed kk=e (pad e>=16 with zero)
  v16bf qf = cat0(ldv8(qbuf + (size_t)g * 256 + h * 16 + kb));

  const bf16_t* krow = kbuf + (size_t)b * SV * 256 + h * 16 + kb;          // + key*256
  const bf16_t* vrow = vt + ((size_t)(b * 256 + h * 16 + ln)) * SV;        // + key

  v8f o = vzero8();
  float m_run = -3.0e38f, l_run = 0.0f;
  int nkt = (q0 >> 4) + 2;                 // keys up to j = q0+31 (mask j<=i+16)

  for (int kt = 0; kt < nkt; ++kt) {
    int k0 = kt * 16;
    if (kt + 1 < nkt)
      __builtin_prefetch(krow + (size_t)(k0 + 16 + ln) * 256, 0, 0);

    // K tile as A fragment: lane=key row, packed kk=e
    v16bf kf = cat0(ldv8(krow + (size_t)(k0 + ln) * 256));
    v8f sc = vzero8();
    sc = wmma_bf16(kf, qf, sc);            // S^T: rows=keys, lanes=queries

    // element r: key j = k0 + 8*hl + r, query i = qi
    float p[8];
    float mloc = -3.0e38f;
#pragma unroll
    for (int r = 0; r < 8; ++r) {
      int j = k0 + kb + r;
      float sv = sc[r] * 0.25f;            // 1/sqrt(16)
      p[r] = ((j - qi) <= VKV) ? sv : -3.0e38f;
      mloc = fmaxf(mloc, p[r]);
    }
    float mtile = fmaxf(mloc, __shfl_xor(mloc, 16, 32));
    float mnew  = fmaxf(m_run, mtile);
    float corr  = __expf(m_run - mnew);

    // exp -> P^T packed directly as B fragment (C rows == B kk packing, no shuffles)
    v16bf pf;
    float lsum = 0.0f;
#pragma unroll
    for (int r = 0; r < 8; ++r) {
      float e = __expf(p[r] - mnew);       // masked -> exp(-huge) = 0
      lsum += e;
      pf[r] = (bf16_t)e;
    }
#pragma unroll
    for (int r = 0; r < 8; ++r) pf[r + 8] = (bf16_t)0.0f;
    lsum += __shfl_xor(lsum, 16, 32);
    l_run = l_run * corr + lsum;
    m_run = mnew;
#pragma unroll
    for (int r = 0; r < 8; ++r) o[r] *= corr;

    // V^T as A fragment: lane=channel row, packed kk=key (contiguous in vt)
    v16bf vf = cat0(ldv8(vrow + k0 + kb));
    o = wmma_bf16(vf, pf, o);              // O^T: rows=channels, lanes=queries
  }

  float inv = 1.0f / l_run;
  v8bf zo;
#pragma unroll
  for (int r = 0; r < 8; ++r) zo[r] = (bf16_t)(o[r] * inv);
  // lane: query g fixed, channels h*16 + 8*hl + r contiguous
  *reinterpret_cast<v8bf*>(zbuf + (size_t)g * 256 + h * 16 + kb) = zo;
}

// ---------------- output projection: C^T = Wo^T(A) x z^T(B), + b_O ----------------
// 256 gtiles x 48 mtiles = 12288 waves

__global__ void __launch_bounds__(256) k_outproj(const bf16_t* __restrict__ zbuf,
                                                 const bf16_t* __restrict__ wot,
                                                 const float* __restrict__ bo,
                                                 float* __restrict__ out) {
  int wid  = (blockIdx.x * blockDim.x + threadIdx.x) >> 5;
  int lane = threadIdx.x & 31;
  int mt = wid % 48;
  int gt = wid / 48;
  int hl = lane >> 4, ln = lane & 15, kb = hl * 8;
  int mA = mt * 16 + ln;          // A-lane output column
  int g  = gt * 16 + ln;          // B-lane row over B*S
  const bf16_t* wrow = wot + (size_t)mA * 256;
  const bf16_t* zrow = zbuf + (size_t)g * 256;

  v8f acc = vzero8();
#pragma unroll
  for (int c0 = 0; c0 < 256; c0 += 32) {
    v16bf a = catv(ldv8(wrow + c0 + kb), ldv8(wrow + c0 + 16 + kb));
    v16bf b = catv(ldv8(zrow + c0 + kb), ldv8(zrow + c0 + 16 + kb));
    acc = wmma_bf16(a, b, acc);
  }
  // C layout: lane -> g (N), vgpr r -> m = mt*16 + 8*hl + r
  int m0 = mt * 16 + 8 * hl;
  float4 lo, hi;
  lo.x = acc[0] + bo[m0 + 0]; lo.y = acc[1] + bo[m0 + 1];
  lo.z = acc[2] + bo[m0 + 2]; lo.w = acc[3] + bo[m0 + 3];
  hi.x = acc[4] + bo[m0 + 4]; hi.y = acc[5] + bo[m0 + 5];
  hi.z = acc[6] + bo[m0 + 6]; hi.w = acc[7] + bo[m0 + 7];
  float* dst = out + (size_t)g * DM + m0;
  *reinterpret_cast<float4*>(dst)     = lo;
  *reinterpret_cast<float4*>(dst + 4) = hi;
}

// ---------------- launch ----------------

extern "C" void kernel_launch(void* const* d_in, const int* in_sizes, int n_in,
                              void* d_out, int out_size, void* d_ws, size_t ws_size,
                              hipStream_t stream) {
  (void)in_sizes; (void)n_in; (void)out_size; (void)ws_size;
  const float* resid = (const float*)d_in[0];
  const float* W_Q   = (const float*)d_in[1];
  const float* W_K   = (const float*)d_in[2];
  const float* W_V   = (const float*)d_in[3];
  const float* W_O   = (const float*)d_in[4];
  const float* b_Q   = (const float*)d_in[5];
  const float* b_K   = (const float*)d_in[6];
  const float* b_V   = (const float*)d_in[7];
  const float* b_O   = (const float*)d_in[8];
  const float* vk    = (const float*)d_in[9];
  const float* vv    = (const float*)d_in[10];
  float* out = (float*)d_out;

  char* ws = (char*)d_ws;
  size_t off = 0;
  auto carve = [&](size_t bytes) -> char* {
    char* p = ws + off;
    off += (bytes + 255) & ~(size_t)255;
    return p;
  };
  bf16_t* xr   = (bf16_t*)carve((size_t)BB * SS * DM * 2);   // resid bf16
  bf16_t* wqk  = (bf16_t*)carve((size_t)512 * DM * 2);       // [q|k] weights, (c,d)
  bf16_t* wv   = (bf16_t*)carve((size_t)256 * DM * 2);       // V weights (n,d)
  bf16_t* wot  = (bf16_t*)carve((size_t)DM * 256 * 2);       // W_O transposed (m,ch)
  bf16_t* qbuf = (bf16_t*)carve((size_t)BB * SS * 256 * 2);  // Q (g, 256)
  bf16_t* kbuf = (bf16_t*)carve((size_t)BB * SV * 256 * 2);  // K (b,pos, 256) incl virtual
  bf16_t* vt   = (bf16_t*)carve((size_t)BB * 256 * SV * 2);  // V channel-major (b,ch,pos)
  bf16_t* zbuf = (bf16_t*)carve((size_t)BB * SS * 256 * 2);  // z (g, 256)

  int n;
  n = BB * SS * DM;
  k_cvt<<<(n + 255) / 256, 256, 0, stream>>>(resid, xr, n);
  n = 256 * DM;
  k_cvt<<<(n + 255) / 256, 256, 0, stream>>>(W_V, wv, n);
  n = 512 * DM;
  k_pack_wqk<<<(n + 255) / 256, 256, 0, stream>>>(W_Q, W_K, wqk, n);
  n = DM * 256;
  k_pack_wo<<<(n + 255) / 256, 256, 0, stream>>>(W_O, wot, n);
  n = BB * VKV * 256;
  k_virtual<<<(n + 255) / 256, 256, 0, stream>>>(vk, vv, kbuf, vt, n);

  k_proj_qk<<<1024, 256, 0, stream>>>(xr, wqk, b_Q, b_K, qbuf, kbuf);  // 8192 waves
  k_proj_v<<<512, 256, 0, stream>>>(xr, wv, b_V, vt);                  // 4096 waves
  k_attn<<<512, 256, 0, stream>>>(qbuf, kbuf, vt, zbuf);               // 4096 waves
  k_outproj<<<1536, 256, 0, stream>>>(zbuf, wot, b_O, out);            // 12288 waves
}
